// AttnDecoderRNN_35613868818797
// MI455X (gfx1250) — matrix operand order repaired and checked by reading, compiled, not run on hardware
//
#include <hip/hip_runtime.h>
#include <hip/hip_bf16.h>

// ---------------------------------------------------------------------------
// AttnDecoderRNN for MI455X (gfx1250): bf16 WMMA GEMMs, fp32 state/softmax.
//   H=1024, V=32000, L=2, B=16, S=200, T=200
// Weights converted to bf16 once (~100 MB -> fits 192 MB L2 for all 200 steps)
// GEMM: one wave computes TWO 16x16 output tiles (shared A fragment, two
// independent accumulator chains), software-pipelined k-loop.
// ---------------------------------------------------------------------------

#define H_  1024
#define V_  32000
#define L_  2
#define B_  16
#define S_  200
#define T_  200
#define G4_ 4096    // 4*H

typedef __attribute__((ext_vector_type(16))) __bf16 v16bf;
typedef __attribute__((ext_vector_type(8)))  __bf16 v8bf;
typedef __attribute__((ext_vector_type(8)))  float  v8f;

__device__ __forceinline__ float sigmoidf_(float x) {
    return 1.0f / (1.0f + __expf(-x));
}

// A fragment 16x32 bf16 (ISA 7.12.2): row M = lane&15;
//   lanes<16: j<8 -> K=kb+j, j>=8 -> K=kb+8+j ; lanes>=16: +8
__device__ __forceinline__ v16bf ldA(const __bf16* __restrict__ p, int kb, int aoff) {
    v8bf lo = *(const v8bf*)(p + kb + aoff);
    v8bf hi = *(const v8bf*)(p + kb + aoff + 16);
    return __builtin_shufflevector(lo, hi,
               0,1,2,3,4,5,6,7,8,9,10,11,12,13,14,15);
}

// B fragment 32x16 bf16: col N = lane&15; same K order as A fragment
__device__ __forceinline__ v16bf ldB(const __bf16* __restrict__ p, int kb, int boff) {
    v8bf lo = *(const v8bf*)(p + kb + boff);
    v8bf hi = *(const v8bf*)(p + kb + boff + 16);
    return __builtin_shufflevector(lo, hi,
               0,1,2,3,4,5,6,7,8,9,10,11,12,13,14,15);
}

// ---------------------------------------------------------------------------
// fp32 -> bf16 conversion (grid-stride)
// ---------------------------------------------------------------------------
__global__ void f2bf_kernel(const float* __restrict__ in, __bf16* __restrict__ out,
                            size_t n) {
    size_t i = (size_t)blockIdx.x * blockDim.x + threadIdx.x;
    size_t stride = (size_t)gridDim.x * blockDim.x;
    for (; i < n; i += stride) out[i] = (__bf16)in[i];
}

// ---------------------------------------------------------------------------
// Embedding gather + bf16 convert:  x[b,k] = emb[tok(b,t), k]
// tokens: t==0 -> SOS(1), else target[b, t-1]
// ---------------------------------------------------------------------------
__global__ void embed_kernel(const int* __restrict__ tgt,
                             const float* __restrict__ emb,
                             __bf16* __restrict__ x, int t) {
    int idx = blockIdx.x * blockDim.x + threadIdx.x;   // B_*H_ threads
    int b = idx >> 10, k = idx & (H_ - 1);
    int tok = (t == 0) ? 1 : tgt[b * T_ + (t - 1)];
    x[idx] = (__bf16)emb[(size_t)tok * H_ + k];
}

// ---------------------------------------------------------------------------
// WMMA GEMM:  out[m,n] = sum_k A1[m,k]*B1[n,k] (+ A2[m,k]*B2[n,k]) + bias(n)
//   A: [16,1024] bf16 row-major,  B: [N,1024] bf16 row-major (x @ W^T)
//   Each wave: two adjacent 16x16 tiles (n0, n0+16); k-loop software-pipelined.
//   C layout: N = lane&15 (+tile), M = r + 8*(lane>=16)
// ---------------------------------------------------------------------------
template <bool DUAL>
__global__ __launch_bounds__(256)
void gemm16_bf16_kernel(const __bf16* __restrict__ A1,
                        const __bf16* __restrict__ A2,
                        const __bf16* __restrict__ B1,
                        const __bf16* __restrict__ B2,
                        const float*  __restrict__ bias1,
                        const float*  __restrict__ bias2,
                        float* __restrict__ out, int N) {
    const int lane = threadIdx.x & 31;
    const int wave = (blockIdx.x * blockDim.x + threadIdx.x) >> 5;
    const int n0   = wave << 5;          // two 16-wide tiles per wave
    if (n0 >= N) return;
    const int nn   = lane & 15;
    const int hi   = lane >> 4;          // 0 or 1
    const int aoff = hi << 3;            // A K-chunk offset (0 or 8)
    const int boff = hi << 3;            // B K-chunk offset (0 or 8)

    const __bf16* a1p  = A1 + (size_t)nn * H_;
    const __bf16* a2p  = DUAL ? (A2 + (size_t)nn * H_) : nullptr;
    const __bf16* b1p0 = B1 + (size_t)(n0 + nn)      * H_;
    const __bf16* b1p1 = B1 + (size_t)(n0 + 16 + nn) * H_;
    const __bf16* b2p0 = DUAL ? (B2 + (size_t)(n0 + nn)      * H_) : nullptr;
    const __bf16* b2p1 = DUAL ? (B2 + (size_t)(n0 + 16 + nn) * H_) : nullptr;

    v8f acc0 = {}, acc1 = {};

    // prologue: fragments for kb = 0
    v16bf a1f  = ldA(a1p, 0, aoff);
    v16bf b10f = ldB(b1p0, 0, boff);
    v16bf b11f = ldB(b1p1, 0, boff);
    v16bf a2f, b20f, b21f;
    if (DUAL) {
        a2f  = ldA(a2p, 0, aoff);
        b20f = ldB(b2p0, 0, boff);
        b21f = ldB(b2p1, 0, boff);
    }

    // pipelined main loop: issue loads for kb+32 before computing kb
    for (int kb = 0; kb < H_ - 32; kb += 32) {
        const int kn = kb + 32;
        __builtin_prefetch(b1p0 + kb + 256, 0, 1);

        v16bf na1  = ldA(a1p, kn, aoff);
        v16bf nb10 = ldB(b1p0, kn, boff);
        v16bf nb11 = ldB(b1p1, kn, boff);
        v16bf na2, nb20, nb21;
        if (DUAL) {
            na2  = ldA(a2p, kn, aoff);
            nb20 = ldB(b2p0, kn, boff);
            nb21 = ldB(b2p1, kn, boff);
        }

        acc0 = __builtin_amdgcn_wmma_f32_16x16x32_bf16(
                   false, a1f, false, b10f, (short)0, acc0, false, false);
        acc1 = __builtin_amdgcn_wmma_f32_16x16x32_bf16(
                   false, a1f, false, b11f, (short)0, acc1, false, false);
        if (DUAL) {
            acc0 = __builtin_amdgcn_wmma_f32_16x16x32_bf16(
                       false, a2f, false, b20f, (short)0, acc0, false, false);
            acc1 = __builtin_amdgcn_wmma_f32_16x16x32_bf16(
                       false, a2f, false, b21f, (short)0, acc1, false, false);
        }

        a1f = na1; b10f = nb10; b11f = nb11;
        if (DUAL) { a2f = na2; b20f = nb20; b21f = nb21; }
    }

    // epilogue: last k-block
    acc0 = __builtin_amdgcn_wmma_f32_16x16x32_bf16(
               false, a1f, false, b10f, (short)0, acc0, false, false);
    acc1 = __builtin_amdgcn_wmma_f32_16x16x32_bf16(
               false, a1f, false, b11f, (short)0, acc1, false, false);
    if (DUAL) {
        acc0 = __builtin_amdgcn_wmma_f32_16x16x32_bf16(
                   false, a2f, false, b20f, (short)0, acc0, false, false);
        acc1 = __builtin_amdgcn_wmma_f32_16x16x32_bf16(
                   false, a2f, false, b21f, (short)0, acc1, false, false);
    }

    float bb0 = bias1 ? bias1[n0 + nn]      : 0.0f;
    float bb1 = bias1 ? bias1[n0 + 16 + nn] : 0.0f;
    if (bias2) { bb0 += bias2[n0 + nn]; bb1 += bias2[n0 + 16 + nn]; }
#pragma unroll
    for (int r = 0; r < 8; ++r) {
        const int m = r + (hi << 3);
        out[(size_t)m * N + n0 + nn]      = acc0[r] + bb0;
        out[(size_t)m * N + n0 + 16 + nn] = acc1[r] + bb1;
    }
}

// ---------------------------------------------------------------------------
// LSTM cell pointwise:  gates [16,4096] (i,f,g,o) -> update h,c (fp32 + bf16 h)
// ---------------------------------------------------------------------------
__global__ void lstm_cell_kernel(const float* __restrict__ gates,
                                 float* __restrict__ h, float* __restrict__ c,
                                 __bf16* __restrict__ h_bf) {
    int idx = blockIdx.x * blockDim.x + threadIdx.x;   // B_*H_
    int b = idx >> 10, j = idx & (H_ - 1);
    const float* g = gates + (size_t)b * G4_;
    float ig = sigmoidf_(g[j]);
    float fg = sigmoidf_(g[H_ + j]);
    float gg = tanhf(g[2 * H_ + j]);
    float og = sigmoidf_(g[3 * H_ + j]);
    float cn = fg * c[idx] + ig * gg;
    float hn = og * tanhf(cn);
    c[idx] = cn;
    h[idx] = hn;
    h_bf[idx] = (__bf16)hn;
}

// ---------------------------------------------------------------------------
// Luong dot attention, one block per batch element (fp32 throughout).
// Writes attn weights to d_out and ctx (fp32 + bf16 for projection GEMM).
// ---------------------------------------------------------------------------
__global__ __launch_bounds__(256)
void attn_kernel(const float* __restrict__ enc,     // [B,S,H]
                 const float* __restrict__ q,       // [B,H] (top layer h)
                 float* __restrict__ ctx,           // [B,H]
                 __bf16* __restrict__ ctx_bf,       // [B,H]
                 float* __restrict__ attn_out) {    // d_out attn base + t*S
    __shared__ float sc[S_];
    __shared__ float red[256];
    const int b = blockIdx.x, tid = threadIdx.x;
    const float* e  = enc + (size_t)b * S_ * H_;
    const float* qb = q + (size_t)b * H_;

    for (int s = tid; s < S_; s += 256) {
        const float* row = e + (size_t)s * H_;
        float acc = 0.0f;
        for (int k = 0; k < H_; ++k) acc += qb[k] * row[k];
        sc[s] = acc;
    }
    __syncthreads();

    float m = -INFINITY;
    for (int s = tid; s < S_; s += 256) m = fmaxf(m, sc[s]);
    red[tid] = m; __syncthreads();
    for (int off = 128; off > 0; off >>= 1) {
        if (tid < off) red[tid] = fmaxf(red[tid], red[tid + off]);
        __syncthreads();
    }
    m = red[0]; __syncthreads();

    float sum = 0.0f;
    for (int s = tid; s < S_; s += 256) {
        float v = __expf(sc[s] - m); sc[s] = v; sum += v;
    }
    red[tid] = sum; __syncthreads();
    for (int off = 128; off > 0; off >>= 1) {
        if (tid < off) red[tid] += red[tid + off];
        __syncthreads();
    }
    sum = red[0]; __syncthreads();

    float inv = 1.0f / sum;
    for (int s = tid; s < S_; s += 256) {
        float a = sc[s] * inv; sc[s] = a;
        attn_out[(size_t)b * (T_ * S_) + s] = a;
    }
    __syncthreads();

    for (int k = tid; k < H_; k += 256) {
        float acc = 0.0f;
        for (int s = 0; s < S_; ++s) acc += sc[s] * e[(size_t)s * H_ + k];
        ctx[(size_t)b * H_ + k]    = acc;
        ctx_bf[(size_t)b * H_ + k] = (__bf16)acc;
    }
}

// ---------------------------------------------------------------------------
// log_softmax over V=32000, one block per batch row. out = d_out lp base + t*V
// ---------------------------------------------------------------------------
__global__ __launch_bounds__(256)
void logsoftmax_kernel(const float* __restrict__ logits,  // [B,V]
                       float* __restrict__ out) {
    __shared__ float red[256];
    const int b = blockIdx.x, tid = threadIdx.x;
    const float* row = logits + (size_t)b * V_;

    float m = -INFINITY;
    for (int v = tid; v < V_; v += 256) m = fmaxf(m, row[v]);
    red[tid] = m; __syncthreads();
    for (int off = 128; off > 0; off >>= 1) {
        if (tid < off) red[tid] = fmaxf(red[tid], red[tid + off]);
        __syncthreads();
    }
    m = red[0]; __syncthreads();

    float sum = 0.0f;
    for (int v = tid; v < V_; v += 256) sum += __expf(row[v] - m);
    red[tid] = sum; __syncthreads();
    for (int off = 128; off > 0; off >>= 1) {
        if (tid < off) red[tid] += red[tid + off];
        __syncthreads();
    }
    float lse = m + __logf(red[0]);

    float* o = out + (size_t)b * ((size_t)T_ * V_);
    for (int v = tid; v < V_; v += 256) o[v] = row[v] - lse;
}

// ---------------------------------------------------------------------------
// Host-side launcher
// ---------------------------------------------------------------------------
extern "C" void kernel_launch(void* const* d_in, const int* in_sizes, int n_in,
                              void* d_out, int out_size, void* d_ws, size_t ws_size,
                              hipStream_t stream) {
    (void)in_sizes; (void)n_in; (void)out_size; (void)ws_size;

    const float* enc   = (const float*)d_in[0];   // [B,S,H]
    const float* h0    = (const float*)d_in[1];   // [L,B,H]
    const float* c0    = (const float*)d_in[2];   // [L,B,H]
    const int*   tgt   = (const int*)  d_in[3];   // [B,T]
    const float* emb   = (const float*)d_in[4];   // [V,H]
    const float* W_ih  = (const float*)d_in[5];   // [L,4H,H]
    const float* W_hh  = (const float*)d_in[6];   // [L,4H,H]
    const float* b_ih  = (const float*)d_in[7];   // [L,4H]
    const float* b_hh  = (const float*)d_in[8];   // [L,4H]
    const float* W_out = (const float*)d_in[9];   // [V,H]
    const float* b_out = (const float*)d_in[10];  // [V]

    float* out = (float*)d_out;
    const size_t LP_OFF = (size_t)B_ * T_ * V_;          // end of log_probs
    const size_t HT_OFF = LP_OFF;                        // hT [L,B,H]
    const size_t CT_OFF = HT_OFF + (size_t)L_ * B_ * H_; // cT
    const size_t AT_OFF = CT_OFF + (size_t)L_ * B_ * H_; // attentions [B,T,S]

    // ---- workspace carve-up (byte offsets, 256-aligned) ----
    char* ws = (char*)d_ws;
    size_t off = 0;
    auto take = [&](size_t bytes) { size_t o = off; off += (bytes + 255) & ~(size_t)255; return o; };
    __bf16* Wih_bf  = (__bf16*)(ws + take((size_t)L_ * G4_ * H_ * 2));  // 16.8 MB
    __bf16* Whh_bf  = (__bf16*)(ws + take((size_t)L_ * G4_ * H_ * 2));  // 16.8 MB
    __bf16* Wout_bf = (__bf16*)(ws + take((size_t)V_ * H_ * 2));        // 65.5 MB
    __bf16* x_bf    = (__bf16*)(ws + take((size_t)B_ * H_ * 2));
    __bf16* h_bf    = (__bf16*)(ws + take((size_t)L_ * B_ * H_ * 2));
    __bf16* ctx_bf  = (__bf16*)(ws + take((size_t)B_ * H_ * 2));
    float*  h_st    = (float*) (ws + take((size_t)L_ * B_ * H_ * 4));
    float*  c_st    = (float*) (ws + take((size_t)L_ * B_ * H_ * 4));
    float*  gates   = (float*) (ws + take((size_t)B_ * G4_ * 4));
    float*  ctx_f   = (float*) (ws + take((size_t)B_ * H_ * 4));
    float*  logits  = (float*) (ws + take((size_t)B_ * V_ * 4));

    // ---- one-time preprocessing: bf16 weight copies + state init ----
    f2bf_kernel<<<2048, 256, 0, stream>>>(W_ih,  Wih_bf,  (size_t)L_ * G4_ * H_);
    f2bf_kernel<<<2048, 256, 0, stream>>>(W_hh,  Whh_bf,  (size_t)L_ * G4_ * H_);
    f2bf_kernel<<<4096, 256, 0, stream>>>(W_out, Wout_bf, (size_t)V_ * H_);
    hipMemcpyAsync(h_st, h0, (size_t)L_ * B_ * H_ * 4, hipMemcpyDeviceToDevice, stream);
    hipMemcpyAsync(c_st, c0, (size_t)L_ * B_ * H_ * 4, hipMemcpyDeviceToDevice, stream);
    f2bf_kernel<<<128, 256, 0, stream>>>(h_st, h_bf, (size_t)L_ * B_ * H_);

    const int BH_BLOCKS = (B_ * H_) / 256;  // 64

    // grid sizes: 2 tiles (32 cols) per wave, 8 waves per block
    const int GATES_BLOCKS = G4_ / 32 / 8;  // 16
    const int PROJ_BLOCKS  = V_  / 32 / 8;  // 125

    // ---- 200 sequential decode steps ----
    for (int t = 0; t < T_; ++t) {
        // 1) token embedding -> x (bf16)
        embed_kernel<<<BH_BLOCKS, 256, 0, stream>>>(tgt, emb, x_bf, t);

        // 2) LSTM layers
        for (int l = 0; l < L_; ++l) {
            const __bf16* a1 = (l == 0) ? x_bf : (h_bf + 0 * B_ * H_);
            gemm16_bf16_kernel<true><<<GATES_BLOCKS, 256, 0, stream>>>(
                a1, h_bf + (size_t)l * B_ * H_,
                Wih_bf + (size_t)l * G4_ * H_, Whh_bf + (size_t)l * G4_ * H_,
                b_ih + (size_t)l * G4_, b_hh + (size_t)l * G4_,
                gates, G4_);
            lstm_cell_kernel<<<BH_BLOCKS, 256, 0, stream>>>(
                gates, h_st + (size_t)l * B_ * H_, c_st + (size_t)l * B_ * H_,
                h_bf + (size_t)l * B_ * H_);
        }

        // 3) attention (query = top-layer h, fp32); writes attn weights to d_out
        attn_kernel<<<B_, 256, 0, stream>>>(
            enc, h_st + (size_t)(L_ - 1) * B_ * H_, ctx_f, ctx_bf,
            out + AT_OFF + (size_t)t * S_);

        // 4) output projection
        gemm16_bf16_kernel<false><<<PROJ_BLOCKS, 256, 0, stream>>>(
            ctx_bf, nullptr, Wout_bf, nullptr, b_out, nullptr, logits, V_);

        // 5) log-softmax -> d_out log_probs[:, t, :]
        logsoftmax_kernel<<<B_, 256, 0, stream>>>(
            logits, out + (size_t)t * V_);
    }

    // ---- final hidden/cell state ----
    hipMemcpyAsync(out + HT_OFF, h_st, (size_t)L_ * B_ * H_ * 4,
                   hipMemcpyDeviceToDevice, stream);
    hipMemcpyAsync(out + CT_OFF, c_st, (size_t)L_ * B_ * H_ * 4,
                   hipMemcpyDeviceToDevice, stream);
}
